// Attention_Joint_MaxPool_36867999269370
// MI455X (gfx1250) — compile-verified, matrix operand order checked by reading, not run on hardware
//
#include <hip/hip_runtime.h>
#include <hip/hip_bf16.h>

typedef __attribute__((ext_vector_type(16))) __bf16 v16bf;
typedef __attribute__((ext_vector_type(8)))  __bf16 v8bf;
typedef __attribute__((ext_vector_type(8)))  float  v8f;

#define Bx 4
#define Cx 512
#define Nx 4096
#define NHx 8
#define HDx 64
#define NKx 1024
#define KRx 2048   // 512 channels * 4 patch positions

// ---------------- packed-B fragment layout ----------------
// For WMMA B operand (32x16 K-major tile), lane l holds col=l&15 and K rows
// q0+{0..7} and q0+16+{0..7} with q0=(l>>4)*8. We store those 16 bf16
// contiguously per lane: idx = ((kt*nTiles + nt)*32 + lane)*16 + i.
__device__ __forceinline__ long pk_idx(int k, int n, int nTiles) {
  int kt = k >> 5, kq = k & 31;
  int nt = n >> 4, col = n & 15;
  int lane = col + ((kq & 8) ? 16 : 0);
  int i = (kq & 16) ? ((kq & 7) + 8) : (kq & 7);
  return ((long)(kt * nTiles + nt) * 32 + lane) * 16 + i;
}

__device__ __forceinline__ v16bf cat8(v8bf lo, v8bf hi) {
  return __builtin_shufflevector(lo, hi, 0,1,2,3,4,5,6,7,8,9,10,11,12,13,14,15);
}

// ---------------- weight prep ----------------

// WqT packed-B: B[c][o] = Wq[o][c];  Wkp[o][c] = Wk[o][c] * bn_scale[c] (row-major A)
__global__ void prep_wqk(const float* __restrict__ Wq, const float* __restrict__ Wk,
                         const float* __restrict__ gamma, const float* __restrict__ var,
                         __bf16* __restrict__ WqT_pk, __bf16* __restrict__ Wkp) {
  int t = blockIdx.x * blockDim.x + threadIdx.x;
  if (t >= Cx * Cx) return;
  int o = t >> 9, c = t & 511;
  WqT_pk[pk_idx(c, o, Cx / 16)] = (__bf16)Wq[t];
  float s = rsqrtf(var[c] + 1e-5f) * gamma[c];
  Wkp[t] = (__bf16)(Wk[t] * s);
}

// Wsr_flat packed-B: B[k=o][n=r] = Wsr[o*2048 + r]
__global__ void prep_wsr(const float* __restrict__ Wsr, __bf16* __restrict__ Wsrf_pk) {
  int t = blockIdx.x * blockDim.x + threadIdx.x;
  if (t >= Cx * KRx) return;
  int o = t >> 11, r = t & (KRx - 1);
  Wsrf_pk[pk_idx(o, r, KRx / 16)] = (__bf16)Wsr[t];
}

// kb[o] = sum_c Wk[o][c] * (bsr[c]*s[c] + beta[c] - mean[c]*s[c])  (one wave per o)
__global__ void prep_kb(const float* __restrict__ Wk, const float* __restrict__ bsr,
                        const float* __restrict__ gamma, const float* __restrict__ beta,
                        const float* __restrict__ mean, const float* __restrict__ var,
                        float* __restrict__ kb) {
  int wave = (blockIdx.x * blockDim.x + threadIdx.x) >> 5;
  int lane = threadIdx.x & 31;
  float sum = 0.f;
  for (int c = lane; c < Cx; c += 32) {
    float s = rsqrtf(var[c] + 1e-5f) * gamma[c];
    float u = bsr[c] * s + beta[c] - mean[c] * s;
    sum += Wk[wave * Cx + c] * u;
  }
  for (int m = 16; m >= 1; m >>= 1) sum += __shfl_xor(sum, m, 32);
  if (lane == 0 && wave < Cx) kb[wave] = sum;
}

// ---------------- data prep ----------------

// xT[b][n][c] = bf16(x[b][c][n])   (row-major A of the q GEMM)
__global__ void pack_xT(const float* __restrict__ x, __bf16* __restrict__ xT) {
  long t = (long)blockIdx.x * blockDim.x + threadIdx.x;
  if (t >= (long)Bx * Cx * Nx) return;
  int n = (int)(t & (Nx - 1));
  int c = (int)((t >> 12) & (Cx - 1));
  int b = (int)(t >> 21);
  xT[((long)b * Nx + n) * Cx + c] = (__bf16)x[t];
}

// patches packed-B: B[k = c*4+p][n = m] from x[b][c][(2i+kh)*64 + (2j+kw)]
__global__ void pack_P(const float* __restrict__ x, __bf16* __restrict__ P_pk) {
  long t = (long)blockIdx.x * blockDim.x + threadIdx.x;
  if (t >= (long)Bx * Cx * NKx) return;
  int m = (int)(t & (NKx - 1));
  int c = (int)((t >> 10) & (Cx - 1));
  int b = (int)(t >> 19);
  int i = m >> 5, j = m & 31;
  const float* src = x + ((long)b * Cx + c) * Nx;
  __bf16* dst = P_pk + (long)b * KRx * NKx;
#pragma unroll
  for (int p = 0; p < 4; ++p) {
    int kh = p >> 1, kw = p & 1;
    dst[pk_idx(c * 4 + p, m, NKx / 16)] = (__bf16)src[(2 * i + kh) * 64 + (2 * j + kw)];
  }
}

// v[b][c] = mean_n x[b][c][n]   (one wave per (b,c))
__global__ void reduce_v(const float* __restrict__ x, float* __restrict__ v) {
  int idx = (blockIdx.x * blockDim.x + threadIdx.x) >> 5;
  int lane = threadIdx.x & 31;
  const float* p = x + (long)idx * Nx;
  float s = 0.f;
  for (int n = lane; n < Nx; n += 32) s += p[n];
  for (int m = 16; m >= 1; m >>= 1) s += __shfl_xor(s, m, 32);
  if (lane == 0) v[idx] = s * (1.0f / Nx);
}

// ---------------- batched bf16 GEMM: C = A * Bpk [+bias] ----------------
// A row-major; B pre-packed in fragment layout; C row-major bf16 or packed-B
// for the next GEMM. Wave computes 32(M) x 64(N); block = 4 waves -> 128x64.
__global__ __launch_bounds__(128) void gemm_bf16(
    const __bf16* __restrict__ A, const __bf16* __restrict__ Bpk, __bf16* __restrict__ C,
    int K, int lda, int nTilesB, int ldc, long sA, long sB, long sC,
    const float* __restrict__ bias, int packC, int nTilesC) {
  int wave = threadIdx.x >> 5;
  int lane = threadIdx.x & 31;
  long zb = blockIdx.z;
  int m0 = blockIdx.y * 128 + wave * 32;
  int n0 = blockIdx.x * 64;
  int nt0 = n0 >> 4;
  const __bf16* Ab = A + zb * sA;
  const __bf16* Bb = Bpk + zb * sB;
  int row = lane & 15;
  int q0 = (lane >> 4) << 3;
  v8f acc[2][4] = {};
  for (int k = 0; k < K; k += 32) {
    int kt = k >> 5;
    v16bf a[2];
#pragma unroll
    for (int mf = 0; mf < 2; ++mf) {
      const __bf16* ap = Ab + (long)(m0 + mf * 16 + row) * lda + k + q0;
      a[mf] = cat8(*(const v8bf*)ap, *(const v8bf*)(ap + 16));
    }
#pragma unroll
    for (int t = 0; t < 4; ++t) {
      v16bf bf = *(const v16bf*)(Bb + ((long)(kt * nTilesB + nt0 + t) * 32 + lane) * 16);
#pragma unroll
      for (int mf = 0; mf < 2; ++mf)
        acc[mf][t] = __builtin_amdgcn_wmma_f32_16x16x32_bf16(
            false, a[mf], false, bf, (short)0, acc[mf][t], false, false);
    }
  }
  int rbase = (lane >> 4) << 3;
  int col = lane & 15;
#pragma unroll
  for (int mf = 0; mf < 2; ++mf)
#pragma unroll
    for (int t = 0; t < 4; ++t)
#pragma unroll
      for (int r = 0; r < 8; ++r) {
        int mrow = m0 + mf * 16 + rbase + r;
        int ncol = n0 + t * 16 + col;
        float vv = acc[mf][t][r] + (bias ? bias[mrow] : 0.0f);
        if (packC) C[zb * sC + pk_idx(mrow, ncol, nTilesC)] = (__bf16)vv;
        else       C[zb * sC + (long)mrow * ldc + ncol] = (__bf16)vv;
      }
}

// ---------------- fused scores + max over keys ----------------
// attnmax[b][h][n] = max_k( 0.125 * sum_d qT[b][n][h*64+d] * kmat[h*64+d][k] )
// kmat is packed-B (nTiles = 64). Wave: 32 token rows x all 1024 keys.
__global__ __launch_bounds__(128) void scores_max(
    const __bf16* __restrict__ qT, const __bf16* __restrict__ kmat_pk,
    float* __restrict__ attnmax) {
  int wave = threadIdx.x >> 5;
  int lane = threadIdx.x & 31;
  int n0 = blockIdx.x * 128 + wave * 32;
  int h = blockIdx.y;
  int b = blockIdx.z;
  int row = lane & 15;
  int q0 = (lane >> 4) << 3;
  const __bf16* Ab = qT + (long)b * Nx * Cx + h * HDx;
  const __bf16* Bb = kmat_pk + (long)b * Cx * NKx;
  v16bf a[2][2];
#pragma unroll
  for (int mf = 0; mf < 2; ++mf) {
    const __bf16* ap = Ab + (long)(n0 + mf * 16 + row) * Cx + q0;
    a[mf][0] = cat8(*(const v8bf*)(ap),      *(const v8bf*)(ap + 16));
    a[mf][1] = cat8(*(const v8bf*)(ap + 32), *(const v8bf*)(ap + 48));
  }
  v8f mx[2];
#pragma unroll
  for (int mf = 0; mf < 2; ++mf)
#pragma unroll
    for (int r = 0; r < 8; ++r) mx[mf][r] = -3.0e38f;
  int kt0 = h * 2;  // packed k-tiles for this head's 64 channel rows
  for (int nt = 0; nt < NKx / 16; ++nt) {
    v16bf b0 = *(const v16bf*)(Bb + ((long)(kt0 * (NKx / 16) + nt) * 32 + lane) * 16);
    v16bf b1 = *(const v16bf*)(Bb + ((long)((kt0 + 1) * (NKx / 16) + nt) * 32 + lane) * 16);
#pragma unroll
    for (int mf = 0; mf < 2; ++mf) {
      v8f acc = {};
      acc = __builtin_amdgcn_wmma_f32_16x16x32_bf16(false, a[mf][0], false, b0, (short)0, acc, false, false);
      acc = __builtin_amdgcn_wmma_f32_16x16x32_bf16(false, a[mf][1], false, b1, (short)0, acc, false, false);
#pragma unroll
      for (int r = 0; r < 8; ++r) mx[mf][r] = fmaxf(mx[mf][r], acc[r]);
    }
  }
  int rbase = (lane >> 4) << 3;
#pragma unroll
  for (int mf = 0; mf < 2; ++mf)
#pragma unroll
    for (int r = 0; r < 8; ++r) {
      float m = mx[mf][r] * 0.125f;  // HD^-0.5
      m = fmaxf(m, __shfl_xor(m, 1, 32));
      m = fmaxf(m, __shfl_xor(m, 2, 32));
      m = fmaxf(m, __shfl_xor(m, 4, 32));
      m = fmaxf(m, __shfl_xor(m, 8, 32));
      if ((lane & 15) == 0)
        attnmax[(((long)b * NHx + h) << 12) + n0 + mf * 16 + rbase + r] = m;
    }
}

// ---------------- rank-1 epilogue pieces ----------------

__global__ void pv_kernel(const float* __restrict__ Wproj, const float* __restrict__ v,
                          float* __restrict__ pv) {
  int idx = (blockIdx.x * blockDim.x + threadIdx.x) >> 5;
  int lane = threadIdx.x & 31;
  int b = idx >> 9, o = idx & 511;
  float s = 0.f;
  for (int c = lane; c < Cx; c += 32) s += Wproj[o * Cx + c] * v[b * Cx + c];
  for (int m = 16; m >= 1; m >>= 1) s += __shfl_xor(s, m, 32);
  if (lane == 0) pv[idx] = s;
}

__global__ void sum_heads(const float* __restrict__ amax, float* __restrict__ asum) {
  int t = blockIdx.x * blockDim.x + threadIdx.x;
  if (t >= Bx * Nx) return;
  int b = t >> 12, n = t & (Nx - 1);
  float s = 0.f;
#pragma unroll
  for (int h = 0; h < NHx; ++h) s += amax[(((long)b * NHx + h) << 12) + n];
  asum[t] = s;
}

__global__ void final_out(const float* __restrict__ pv, const float* __restrict__ asum,
                          const float* __restrict__ bproj, float* __restrict__ out) {
  long t = (long)blockIdx.x * blockDim.x + threadIdx.x;
  if (t >= (long)Bx * Cx * Nx) return;
  int n = (int)(t & (Nx - 1));
  int o = (int)((t >> 12) & (Cx - 1));
  int b = (int)(t >> 21);
  out[t] = pv[b * Cx + o] * asum[b * Nx + n] + bproj[o];
}

// ---------------- launch ----------------

extern "C" void kernel_launch(void* const* d_in, const int* in_sizes, int n_in,
                              void* d_out, int out_size, void* d_ws, size_t ws_size,
                              hipStream_t stream) {
  const float* x     = (const float*)d_in[0];
  const float* Wq    = (const float*)d_in[2];
  const float* Wk    = (const float*)d_in[3];
  const float* Wsr   = (const float*)d_in[4];
  const float* bsr   = (const float*)d_in[5];
  const float* gamma = (const float*)d_in[6];
  const float* beta  = (const float*)d_in[7];
  const float* mean  = (const float*)d_in[8];
  const float* var   = (const float*)d_in[9];
  const float* Wproj = (const float*)d_in[10];
  const float* bproj = (const float*)d_in[11];
  float* out = (float*)d_out;

  size_t off = 0;
  auto carve = [&](size_t bytes) {
    void* p = (char*)d_ws + off;
    off += (bytes + 255) & ~(size_t)255;
    return p;
  };
  __bf16* xT    = (__bf16*)carve((size_t)Bx * Nx * Cx * 2);   // 16 MB, row-major A
  __bf16* P_pk  = (__bf16*)carve((size_t)Bx * KRx * NKx * 2); // 16 MB, packed B
  __bf16* qT    = (__bf16*)carve((size_t)Bx * Nx * Cx * 2);   // 16 MB, row-major A
  __bf16* WqTpk = (__bf16*)carve((size_t)Cx * Cx * 2);
  __bf16* Wkp   = (__bf16*)carve((size_t)Cx * Cx * 2);
  __bf16* Wsrpk = (__bf16*)carve((size_t)Cx * KRx * 2);
  __bf16* Wck   = (__bf16*)carve((size_t)Cx * KRx * 2);       // row-major A
  __bf16* km_pk = (__bf16*)carve((size_t)Bx * Cx * NKx * 2);  // packed B
  float* kb    = (float*)carve((size_t)Cx * 4);
  float* vbuf  = (float*)carve((size_t)Bx * Cx * 4);
  float* pv    = (float*)carve((size_t)Bx * Cx * 4);
  float* amax  = (float*)carve((size_t)Bx * NHx * Nx * 4);
  float* asum  = (float*)carve((size_t)Bx * Nx * 4);
  (void)ws_size; (void)in_sizes; (void)n_in; (void)out_size;

  prep_wqk<<<(Cx * Cx) / 256, 256, 0, stream>>>(Wq, Wk, gamma, var, WqTpk, Wkp);
  prep_wsr<<<(Cx * KRx) / 256, 256, 0, stream>>>(Wsr, Wsrpk);
  prep_kb<<<Cx / 8, 256, 0, stream>>>(Wk, bsr, gamma, beta, mean, var, kb);
  pack_xT<<<(Bx * Cx * Nx) / 256, 256, 0, stream>>>(x, xT);
  pack_P<<<(Bx * Cx * NKx) / 256, 256, 0, stream>>>(x, P_pk);
  reduce_v<<<(Bx * Cx) / 8, 256, 0, stream>>>(x, vbuf);

  // qT[b] (4096x512) = xT[b] * WqT   (row-major out: A of scores)
  gemm_bf16<<<dim3(Cx / 64, Nx / 128, Bx), 128, 0, stream>>>(
      xT, WqTpk, qT, Cx, Cx, Cx / 16, Cx, (long)Nx * Cx, 0, (long)Nx * Cx,
      nullptr, 0, 0);
  // Wck (512x2048) = Wkp * Wsr_flat  (row-major out: A of kmat GEMM)
  gemm_bf16<<<dim3(KRx / 64, Cx / 128, 1), 128, 0, stream>>>(
      Wkp, Wsrpk, Wck, Cx, Cx, KRx / 16, KRx, 0, 0, 0, nullptr, 0, 0);
  // kmat[b] (512x1024) = Wck * P[b] + kb  (PACKED out: B of scores)
  gemm_bf16<<<dim3(NKx / 64, Cx / 128, Bx), 128, 0, stream>>>(
      Wck, P_pk, km_pk, KRx, KRx, NKx / 16, 0, 0, (long)KRx * NKx, (long)Cx * NKx,
      kb, 1, NKx / 16);

  scores_max<<<dim3(Nx / 128, NHx, Bx), 128, 0, stream>>>(qT, km_pk, amax);

  pv_kernel<<<(Bx * Cx) / 8, 256, 0, stream>>>(Wproj, vbuf, pv);
  sum_heads<<<(Bx * Nx) / 256, 256, 0, stream>>>(amax, asum);
  final_out<<<(Bx * Cx * Nx) / 256, 256, 0, stream>>>(pv, asum, bproj, out);
}
